// Spectral_Spatial_Modulation_Unit_15848429322481
// MI455X (gfx1250) — matrix operand order repaired and checked by reading, not compiled, not run on hardware
//
#include <hip/hip_runtime.h>
#include <hip/hip_bf16.h>

// ---------------------------------------------------------------------------
// Spectral Spatial Modulation Unit — gfx1250 (MI455X), wave32, WMMA f16->f32
// ---------------------------------------------------------------------------

#define DIM   96
#define NHEAD 4
#define CHH   24
#define WIN   8
#define MIDC  48
#define BN    2
#define HH    256
#define WW    256
#define WF    129            // WW/2 + 1
#define HW    (HH*WW)        // 65536
#define HWF   (HH*WF)        // 33024
#define NPIX  ((long)BN*DIM*HW)
#define NF    ((long)BN*DIM*HWF)

typedef __attribute__((ext_vector_type(16))) _Float16 v16h;
typedef __attribute__((ext_vector_type(8)))  float    v8f;

__device__ __forceinline__ float gelu_f(float v) {
    return 0.5f * v * (1.0f + erff(v * 0.7071067811865475f));
}

// ---- WMMA operand gathers (wave32; A and B both use k = 16*(lane/16) + s) ----------------------
// A: 16x32 f16, row-major weight tile in LDS, W[m][k] with row stride Ks.
__device__ __forceinline__ v16h loadA(const _Float16* __restrict__ Wl, int Ks,
                                      int m0, int k0, int lane) {
    int m  = m0 + (lane & 15);
    int kh = (lane & 16) ? 16 : 0;
    v16h a;
#pragma unroll
    for (int v = 0; v < 8; ++v) {
        int kb = k0 + 2*v + kh;
        a[2*v]   = Wl[m*Ks + kb];
        a[2*v+1] = Wl[m*Ks + kb + 1];
    }
    return a;
}
// B: 32x16 f16, activation tile in LDS, X[p][k] (pixel-major, row stride Ks).
__device__ __forceinline__ v16h loadB(const _Float16* __restrict__ Xl, int Ks,
                                      int p0, int k0, int lane) {
    int p  = p0 + (lane & 15);
    int kg = k0 + ((lane & 16) ? 16 : 0);
    v16h b;
#pragma unroll
    for (int v = 0; v < 8; ++v) {
        b[2*v]   = Xl[p*Ks + kg + 2*v];
        b[2*v+1] = Xl[p*Ks + kg + 2*v + 1];
    }
    return b;
}
#define WMMA_F16(A_, B_, C_) \
    __builtin_amdgcn_wmma_f32_16x16x32_f16(false, (A_), false, (B_), (short)0, (C_), false, false)

// ---------------------------------------------------------------------------
// channel-first LayerNorm: xn = lw*(x-mu)/sqrt(var+eps) + lb
// ---------------------------------------------------------------------------
__global__ __launch_bounds__(256) void k_ln(const float* __restrict__ x,
                                            const float* __restrict__ lw,
                                            const float* __restrict__ lb,
                                            float* __restrict__ xn) {
    long p = (long)blockIdx.x * 256 + threadIdx.x;   // pixel id in [0, BN*HW)
    int  b = (int)(p >> 16);
    long pp = p & 65535;
    const float* base = x + (((long)b * DIM) << 16) + pp;
    float s = 0.f, ss = 0.f;
#pragma unroll 4
    for (int c = 0; c < DIM; ++c) {
        float v = base[(long)c << 16];
        s += v; ss += v * v;
    }
    float mu  = s * (1.0f / DIM);
    float var = ss * (1.0f / DIM) - mu * mu;
    float inv = rsqrtf(var + 1e-6f);
    float* ob = xn + (((long)b * DIM) << 16) + pp;
#pragma unroll 4
    for (int c = 0; c < DIM; ++c) {
        float v = base[(long)c << 16];
        ob[(long)c << 16] = lw[c] * (v - mu) * inv + lb[c];
    }
}

// ---------------------------------------------------------------------------
// depthwise 3x3, SAME(zero) padding, optional bias / gelu
// ---------------------------------------------------------------------------
template<bool BIAS, bool GELU>
__global__ __launch_bounds__(256) void k_dw3(const float* __restrict__ in,
                                             const float* __restrict__ k,
                                             const float* __restrict__ bias,
                                             float* __restrict__ out) {
    long i = (long)blockIdx.x * 256 + threadIdx.x;   // element id in [0, NPIX)
    int  wc = (int)(i & 255);
    long t  = i >> 8;
    int  h  = (int)(t & 255);
    long bc = t >> 8;                                 // b*96 + c
    int  c  = (int)(bc % DIM);
    const float* base = in + (bc << 16);
    float acc = 0.f;
#pragma unroll
    for (int dy = -1; dy <= 1; ++dy) {
        int hh = h + dy;
        if ((unsigned)hh >= (unsigned)HH) continue;
#pragma unroll
        for (int dx = -1; dx <= 1; ++dx) {
            int ww2 = wc + dx;
            if ((unsigned)ww2 >= (unsigned)WW) continue;
            acc += base[hh*WW + ww2] * k[c*9 + (dy+1)*3 + (dx+1)];
        }
    }
    if (BIAS) acc += bias[c];
    if (GELU) acc = gelu_f(acc);
    out[i] = acc;
}

// ---------------------------------------------------------------------------
// generic WMMA 1x1 conv over pixels: out[o,p] = act(W[o,:] . in[:,p] (+bias))
//   SPLITIN : input channels come half from in0, half from in1 (cat)
//   SPLITOUT: output channels split to out0/out1
//   INH     : input tensor is f16 (Hb buffer)
//   OUTH    : output written as f16 to outh
// in-place (out0 == in0) is safe: each block stages its pixels to LDS first.
// ---------------------------------------------------------------------------
template<int CIN, int COUT, int TPIX, bool SPLITIN, bool SPLITOUT,
         bool INH, bool OUTH, bool ACTGELU, bool BIAS>
__global__ __launch_bounds__(TPIX*2) void k_conv1x1(
        const void* in0_, const void* in1_,
        const float* __restrict__ w, const float* __restrict__ bias,
        float* __restrict__ out0, float* __restrict__ out1,
        _Float16* __restrict__ outh, int pixPerBatch) {
    __shared__ _Float16 sW[COUT*CIN];
    __shared__ _Float16 sX[TPIX*CIN];
    const int tid  = threadIdx.x;
    const int nthr = TPIX*2;

    for (int i = tid; i < COUT*CIN; i += nthr) sW[i] = (_Float16)w[i];

    long q0 = (long)blockIdx.x * TPIX;        // tiles never straddle batches
    int  b  = (int)(q0 / pixPerBatch);
    long p0 = q0 - (long)b * pixPerBatch;
    const int CHALF = SPLITIN ? CIN/2 : CIN;

    for (int i = tid; i < TPIX*CIN; i += nthr) {
        int c = i / TPIX, pp = i % TPIX;
        if (INH) {
            const _Float16* inh = (const _Float16*)in0_;
            sX[pp*CIN + c] = inh[((long)b*CIN + c)*pixPerBatch + p0 + pp];
        } else {
            bool hi = SPLITIN && (c >= CHALF);
            const float* src = (const float*)(hi ? in1_ : in0_);
            int cc = hi ? c - CHALF : c;
            sX[pp*CIN + c] =
                (_Float16)src[((long)b*CHALF + cc)*pixPerBatch + p0 + pp];
        }
    }
    __syncthreads();

    int wave = tid >> 5, lane = tid & 31;
    int pw = wave * 16;
    int pl = pw + (lane & 15);
#pragma unroll
    for (int ot = 0; ot < COUT/16; ++ot) {
        v8f acc = {0.f,0.f,0.f,0.f,0.f,0.f,0.f,0.f};
#pragma unroll
        for (int ks = 0; ks < CIN/32; ++ks) {
            v16h A  = loadA(sW, CIN, ot*16, ks*32, lane);
            v16h Bm = loadB(sX, CIN, pw,    ks*32, lane);
            acc = WMMA_F16(A, Bm, acc);
        }
#pragma unroll
        for (int r = 0; r < 8; ++r) {
            int c = ot*16 + r + ((lane & 16) ? 8 : 0);
            float v = acc[r];
            if (BIAS)    v += bias[c];
            if (ACTGELU) v = gelu_f(v);
            long p = p0 + pl;
            if (OUTH) {
                outh[((long)b*COUT + c)*pixPerBatch + p] = (_Float16)v;
            } else if (SPLITOUT) {
                const int oh = COUT/2;
                if (c >= oh) out1[((long)b*oh + (c-oh))*pixPerBatch + p] = v;
                else         out0[((long)b*oh + c)*pixPerBatch + p] = v;
            } else {
                out0[((long)b*COUT + c)*pixPerBatch + p] = v;
            }
        }
    }
}

// ---------------------------------------------------------------------------
// fused spectral gate: g = sigmoid(W2 gelu(W1 |Xf| + b1) + b2); Xf *= g
// hidden K padded 48 -> 64 with zeros for the 16x16x32 WMMA.
// ---------------------------------------------------------------------------
__global__ __launch_bounds__(256) void k_gate(float* __restrict__ Fr,
                                              float* __restrict__ Fi,
                                              const float* __restrict__ w1,
                                              const float* __restrict__ b1,
                                              const float* __restrict__ w2,
                                              const float* __restrict__ b2) {
    __shared__ _Float16 sW1[MIDC*DIM];      //  9 KB
    __shared__ _Float16 sW2[DIM*64];        // 12 KB (K padded to 64)
    __shared__ _Float16 sM [128*DIM];       // 24 KB
    __shared__ _Float16 sH [128*64];        // 16 KB
    int tid = threadIdx.x;

    for (int i = tid; i < MIDC*DIM; i += 256) sW1[i] = (_Float16)w1[i];
    for (int i = tid; i < DIM*64; i += 256) {
        int o = i >> 6, kk = i & 63;
        sW2[i] = (_Float16)((kk < MIDC) ? w2[o*MIDC + kk] : 0.0f);
    }
    long q0 = (long)blockIdx.x * 128;
    int  b  = (int)(q0 / HWF);
    long p0 = q0 - (long)b * HWF;
    for (int i = tid; i < 128*DIM; i += 256) {
        int c = i >> 7, pp = i & 127;
        long idx = ((long)b*DIM + c)*HWF + p0 + pp;
        float xr = Fr[idx], xi = Fi[idx];
        sM[pp*DIM + c] = (_Float16)sqrtf(xr*xr + xi*xi);
    }
    for (int i = tid; i < 128*16; i += 256) {     // zero the K pad
        int pp = i >> 4;
        sH[pp*64 + MIDC + (i & 15)] = (_Float16)0.0f;
    }
    __syncthreads();

    int wave = tid >> 5, lane = tid & 31;
    int pw = wave * 16, pl = pw + (lane & 15);
#pragma unroll
    for (int ot = 0; ot < MIDC/16; ++ot) {        // 48 hidden = 3 tiles
        v8f acc = {0.f,0.f,0.f,0.f,0.f,0.f,0.f,0.f};
#pragma unroll
        for (int ks = 0; ks < DIM/32; ++ks) {
            v16h A  = loadA(sW1, DIM, ot*16, ks*32, lane);
            v16h Bm = loadB(sM,  DIM, pw,    ks*32, lane);
            acc = WMMA_F16(A, Bm, acc);
        }
#pragma unroll
        for (int r = 0; r < 8; ++r) {
            int c = ot*16 + r + ((lane & 16) ? 8 : 0);
            sH[pl*64 + c] = (_Float16)gelu_f(acc[r] + b1[c]);
        }
    }
    __syncthreads();
#pragma unroll
    for (int ot = 0; ot < DIM/16; ++ot) {
        v8f acc = {0.f,0.f,0.f,0.f,0.f,0.f,0.f,0.f};
#pragma unroll
        for (int ks = 0; ks < 2; ++ks) {
            v16h A  = loadA(sW2, 64, ot*16, ks*32, lane);
            v16h Bm = loadB(sH,  64, pw,    ks*32, lane);
            acc = WMMA_F16(A, Bm, acc);
        }
#pragma unroll
        for (int r = 0; r < 8; ++r) {
            int c = ot*16 + r + ((lane & 16) ? 8 : 0);
            float g = 1.0f / (1.0f + expf(-(acc[r] + b2[c])));
            long idx = ((long)b*DIM + c)*HWF + p0 + pl;
            Fr[idx] *= g;
            Fi[idx] *= g;
        }
    }
}

// ---------------------------------------------------------------------------
// LDS radix-2 DIT FFT-256, 64 threads per transform (bit-reversed input)
// ---------------------------------------------------------------------------
__device__ __forceinline__ void fft256(float2* s, int t, float sign) {
#pragma unroll
    for (int len = 2; len <= 256; len <<= 1) {
        __syncthreads();
        const int half = len >> 1;
#pragma unroll
        for (int u = 0; u < 2; ++u) {
            int j   = t + (u << 6);
            int grp = j / half;
            int pos = j - grp * half;
            int i0  = grp * len + pos;
            int i1  = i0 + half;
            float ang = sign * 6.283185307179586f * (float)pos / (float)len;
            float sn, cs;
            __sincosf(ang, &sn, &cs);
            float2 a = s[i0], bv = s[i1];
            float rx = bv.x*cs - bv.y*sn;
            float ry = bv.x*sn + bv.y*cs;
            s[i0] = make_float2(a.x + rx, a.y + ry);
            s[i1] = make_float2(a.x - rx, a.y - ry);
        }
    }
    __syncthreads();
}

// rows: real 256 -> 129 bins (ortho share 1/16)
__global__ __launch_bounds__(256) void k_rfft_row(const float* __restrict__ xn,
                                                  float* __restrict__ Fr,
                                                  float* __restrict__ Fi) {
    __shared__ float2 sd[4][256];
    int tid = threadIdx.x, r = tid >> 6, t = tid & 63;
    long row = (long)blockIdx.x * 4 + r;       // (b*96+c)*256 + h
    float2* s = sd[r];
    const float* src = xn + (row << 8);
#pragma unroll
    for (int k = 0; k < 4; ++k) {
        int j  = t + (k << 6);
        int br = (int)(__brev((unsigned)j) >> 24);
        s[br] = make_float2(src[j], 0.0f);
    }
    fft256(s, t, -1.0f);
    long bc = row >> 8;
    int  h  = (int)(row & 255);
    long ob = bc * HWF + (long)h * WF;
    for (int wf = t; wf < WF; wf += 64) {
        Fr[ob + wf] = s[wf].x * 0.0625f;
        Fi[ob + wf] = s[wf].y * 0.0625f;
    }
}

// columns: complex 256 in-place, sign = -1 fwd / +1 inv, x 1/16
__global__ __launch_bounds__(256) void k_fft_col(float* __restrict__ Fr,
                                                 float* __restrict__ Fi,
                                                 float sign) {
    __shared__ float2 sd[4][256];
    int tid = threadIdx.x, r = tid >> 6, t = tid & 63;
    long col = (long)blockIdx.x * 4 + r;       // bc*129 + wf
    long bc  = col / WF;
    int  wf  = (int)(col - bc * WF);
    long base = bc * HWF + wf;
    float2* s = sd[r];
#pragma unroll
    for (int k = 0; k < 4; ++k) {
        int j  = t + (k << 6);
        int br = (int)(__brev((unsigned)j) >> 24);
        s[br] = make_float2(Fr[base + (long)j*WF], Fi[base + (long)j*WF]);
    }
    fft256(s, t, sign);
#pragma unroll
    for (int k = 0; k < 4; ++k) {
        int j = t + (k << 6);
        Fr[base + (long)j*WF] = s[j].x * 0.0625f;
        Fi[base + (long)j*WF] = s[j].y * 0.0625f;
    }
}

// rows inverse: 129 bins -> real 256 via Hermitian reconstruction
__global__ __launch_bounds__(256) void k_irfft_row(const float* __restrict__ Fr,
                                                   const float* __restrict__ Fi,
                                                   float* __restrict__ outp) {
    __shared__ float2 sd[4][256];
    int tid = threadIdx.x, r = tid >> 6, t = tid & 63;
    long row = (long)blockIdx.x * 4 + r;
    long bc  = row >> 8;
    int  h   = (int)(row & 255);
    long ib  = bc * HWF + (long)h * WF;
    float2* s = sd[r];
#pragma unroll
    for (int k = 0; k < 4; ++k) {
        int j   = t + (k << 6);
        int src = (j <= 128) ? j : 256 - j;
        float sr = Fr[ib + src];
        float si = (j > 128) ? -Fi[ib + src] : Fi[ib + src];
        int br = (int)(__brev((unsigned)j) >> 24);
        s[br] = make_float2(sr, si);
    }
    fft256(s, t, +1.0f);
    long ob = row << 8;
#pragma unroll
    for (int k = 0; k < 4; ++k) {
        int j = t + (k << 6);
        outp[ob + j] = s[j].x * 0.0625f;
    }
}

// ---------------------------------------------------------------------------
// windowed gated softmax over 8x8 tokens: one wave per window, in-place on f
// ---------------------------------------------------------------------------
__global__ __launch_bounds__(256) void k_softmax(float* __restrict__ f,
                                                 const float* __restrict__ sb,
                                                 const float* __restrict__ temp) {
    int tid = threadIdx.x, wave = tid >> 5, lane = tid & 31;
    long win = (long)blockIdx.x * 8 + wave;    // ((b*96+c)*32 + hb)*32 + wb
    int  wb  = (int)(win & 31);
    long t2  = win >> 5;
    int  hb  = (int)(t2 & 31);
    long bc  = t2 >> 5;
    int  c   = (int)(bc % DIM);
    float tp = temp[c / CHH];
    float tv = log1pf(expf(tp));               // softplus
    long base = (bc << 16) + (long)hb * 8 * WW + (long)wb * 8;

    int e0 = lane,      d0y = e0 >> 3, d0x = e0 & 7;
    int e1 = lane + 32, d1y = e1 >> 3, d1x = e1 & 7;
    long i0 = base + d0y*WW + d0x;
    long i1 = base + d1y*WW + d1x;
    float v0 = f[i0] * sb[i0] * tv;
    float v1 = f[i1] * sb[i1] * tv;

    float m = fmaxf(v0, v1);
#pragma unroll
    for (int off = 16; off; off >>= 1) m = fmaxf(m, __shfl_xor(m, off));
    float ex0 = expf(v0 - m), ex1 = expf(v1 - m);
    float ssum = ex0 + ex1;
#pragma unroll
    for (int off = 16; off; off >>= 1) ssum += __shfl_xor(ssum, off);
    float inv = 1.0f / ssum;
    f[i0] = ex0 * inv;
    f[i1] = ex1 * inv;
}

// ---------------------------------------------------------------------------
// final: y = po_w @ (attn + pos) + po_b ; out = ch_scale*y + x   (WMMA)
// ---------------------------------------------------------------------------
__global__ __launch_bounds__(256) void k_final(const float* __restrict__ attn,
                                               const float* __restrict__ pos,
                                               const float* __restrict__ w,
                                               const float* __restrict__ bias,
                                               const float* __restrict__ scl,
                                               const float* __restrict__ x,
                                               float* __restrict__ outp) {
    __shared__ _Float16 sW[DIM*DIM];
    __shared__ _Float16 sX[128*DIM];
    int tid = threadIdx.x;
    for (int i = tid; i < DIM*DIM; i += 256) sW[i] = (_Float16)w[i];
    long q0 = (long)blockIdx.x * 128;
    int  b  = (int)(q0 >> 16);
    long p0 = q0 & 65535;
    for (int i = tid; i < 128*DIM; i += 256) {
        int c = i >> 7, pp = i & 127;
        long idx = (((long)b*DIM + c) << 16) + p0 + pp;
        sX[pp*DIM + c] = (_Float16)(attn[idx] + pos[idx]);
    }
    __syncthreads();
    int wave = tid >> 5, lane = tid & 31;
    int pw = wave * 16, pl = pw + (lane & 15);
#pragma unroll
    for (int ot = 0; ot < DIM/16; ++ot) {
        v8f acc = {0.f,0.f,0.f,0.f,0.f,0.f,0.f,0.f};
#pragma unroll
        for (int ks = 0; ks < DIM/32; ++ks) {
            v16h A  = loadA(sW, DIM, ot*16, ks*32, lane);
            v16h Bm = loadB(sX, DIM, pw,    ks*32, lane);
            acc = WMMA_F16(A, Bm, acc);
        }
#pragma unroll
        for (int r = 0; r < 8; ++r) {
            int c = ot*16 + r + ((lane & 16) ? 8 : 0);
            float y = acc[r] + bias[c];
            long idx = (((long)b*DIM + c) << 16) + p0 + pl;
            outp[idx] = scl[c] * y + x[idx];
        }
    }
}

// ---------------------------------------------------------------------------
extern "C" void kernel_launch(void* const* d_in, const int* in_sizes, int n_in,
                              void* d_out, int out_size, void* d_ws, size_t ws_size,
                              hipStream_t stream) {
    const float* x           = (const float*)d_in[0];
    const float* ln_w        = (const float*)d_in[1];
    const float* ln_b        = (const float*)d_in[2];
    const float* sg_proj_in  = (const float*)d_in[3];   // [96][192]
    const float* sg_proj_out = (const float*)d_in[4];   // [192][96]
    const float* sg_mg_w1    = (const float*)d_in[5];   // [48][96]
    const float* sg_mg_b1    = (const float*)d_in[6];
    const float* sg_mg_w2    = (const float*)d_in[7];   // [96][48]
    const float* sg_mg_b2    = (const float*)d_in[8];
    const float* v_w1        = (const float*)d_in[9];   // [96][96]
    const float* v_dw        = (const float*)d_in[10];  // [96][9]
    const float* v_w2        = (const float*)d_in[11];
    const float* temp        = (const float*)d_in[12];
    const float* pos_w       = (const float*)d_in[13];
    const float* pos_b       = (const float*)d_in[14];
    const float* po_w        = (const float*)d_in[15];
    const float* po_b        = (const float*)d_in[16];
    const float* ch_scale    = (const float*)d_in[17];
    float* outp = (float*)d_out;

    float* wsf = (float*)d_ws;
    float* xn  = wsf;                 // [BN,96,H,W]   xnorm
    float* pos = xn + NPIX;           // pos_enc
    float* T   = pos + NPIX;          // v1 -> f (irfft) -> attn
    float* U   = T + NPIX;            // dwgelu -> sb
    float* Fr  = U + NPIX;            // [BN,96,H,WF]
    float* Fi  = Fr + NF;
    _Float16* Hb = (_Float16*)(Fi + NF);   // proj hidden (f16)

    dim3 b256(256), b128(128);

    // 1) layernorm + positional dwconv
    k_ln<<<dim3((unsigned)(BN*HW/256)), b256, 0, stream>>>(x, ln_w, ln_b, xn);
    k_dw3<true, false><<<dim3((unsigned)(NPIX/256)), b256, 0, stream>>>(x, pos_w, pos_b, pos);

    // 2) spatial branch: v1 -> dw+gelu -> v2 (in-place)
    k_conv1x1<96,96,128,false,false,false,false,false,false>
        <<<dim3((unsigned)(BN*HW/128)), b256, 0, stream>>>(xn, nullptr, v_w1, nullptr, T, nullptr, nullptr, HW);
    k_dw3<false, true><<<dim3((unsigned)(NPIX/256)), b256, 0, stream>>>(T, v_dw, nullptr, U);
    k_conv1x1<96,96,128,false,false,false,false,false,false>
        <<<dim3((unsigned)(BN*HW/128)), b256, 0, stream>>>(U, nullptr, v_w2, nullptr, U, nullptr, nullptr, HW);

    // 3) spectral branch: rfft2
    k_rfft_row<<<dim3((unsigned)(BN*DIM*HH/4)), b256, 0, stream>>>(xn, Fr, Fi);
    k_fft_col<<<dim3((unsigned)(BN*DIM*WF/4)), b256, 0, stream>>>(Fr, Fi, -1.0f);

    // 4) magnitude gate (fused MLP + sigmoid + complex scale)
    k_gate<<<dim3((unsigned)(BN*HWF/128)), b256, 0, stream>>>(Fr, Fi, sg_mg_w1, sg_mg_b1, sg_mg_w2, sg_mg_b2);

    // 5) complex channel-mix projection: 192 -> 96(gelu, f16) -> 192
    k_conv1x1<192,96,64,true,false,false,true,true,false>
        <<<dim3((unsigned)(BN*HWF/64)), b128, 0, stream>>>(Fr, Fi, sg_proj_in, nullptr, nullptr, nullptr, Hb, HWF);
    k_conv1x1<96,192,128,false,true,true,false,false,false>
        <<<dim3((unsigned)(BN*HWF/128)), b256, 0, stream>>>(Hb, nullptr, sg_proj_out, nullptr, Fr, Fi, nullptr, HWF);

    // 6) irfft2 -> f (into T)
    k_fft_col<<<dim3((unsigned)(BN*DIM*WF/4)), b256, 0, stream>>>(Fr, Fi, +1.0f);
    k_irfft_row<<<dim3((unsigned)(BN*DIM*HH/4)), b256, 0, stream>>>(Fr, Fi, T);

    // 7) windowed gated softmax (in-place on T), then fused output projection
    k_softmax<<<dim3((unsigned)((long)BN*DIM*32*32/8)), b256, 0, stream>>>(T, U, temp);
    k_final<<<dim3((unsigned)(BN*HW/128)), b256, 0, stream>>>(T, pos, po_w, po_b, ch_scale, x, outp);
}
